// WSDA_85375359910386
// MI455X (gfx1250) — matrix operand order repaired and checked by reading, not compiled, not run on hardware
//
#include <hip/hip_runtime.h>
#include <hip/hip_bf16.h>

// ---------------------------------------------------------------------------
// WSDA forward for MI455X (gfx1250, wave32).
//  * All GEMM-shaped work on v_wmma_f32_16x16x32_f16 (f16 in, f32 acc).
//  * GEMM B tiles staged to LDS with the Tensor Data Mover
//    (tensor_load_to_lds, TENSORcnt double buffering) and shared by all
//    4 waves of the block; 32 M-rows per wave reuse each B element twice.
//  * 3x3 conv -> im2col GEMM.  Wavelet cross-attention -> flash attention
//    (WMMA QK^T and PV, online softmax, P transposed C->A layout via LDS).
// ---------------------------------------------------------------------------

typedef __attribute__((ext_vector_type(16))) _Float16 v16h;
typedef __attribute__((ext_vector_type(8)))  _Float16 v8h;
typedef __attribute__((ext_vector_type(8)))  float    v8f;
typedef __attribute__((ext_vector_type(4)))  unsigned int v4u;
typedef __attribute__((ext_vector_type(8)))  int      v8i;
typedef __attribute__((ext_vector_type(4)))  int      v4i;

union HVec16 { v16h v; v8h half8[2]; _Float16 e[16]; };

#define B_     4
#define HH_    64
#define WW_    64
#define NTOK   4096
#define DIM_   512
#define SDIM_  256
#define WDIM_  256
#define RED_   64
#define HD_    64
#define KVLEN  1024
#define SCALE_ 0.125f
#define EPS_   1e-5f

__device__ __forceinline__ v8f wmma_acc(v16h a, v16h b, v8f c) {
  // (neg_a, A, neg_b, B, c_mod, C, reuse_a, reuse_b)
  return __builtin_amdgcn_wmma_f32_16x16x32_f16(false, a, false, b, (short)0, c, false, false);
}

__device__ __forceinline__ v8f zero8() {
  v8f z;
#pragma unroll
  for (int i = 0; i < 8; ++i) z[i] = 0.0f;
  return z;
}

// ---------------------------------------------------------------------------
// TDM: DMA a 2-D tile of 16-bit elements (rows x cols, row stride in elems)
// from global memory into LDS.  Descriptor per CDNA5 ISA 8.3/8.4:
//   group0: count=1 | lds_addr | global_addr[56:0] | type=2
//   group1: data_size=2B, tensor_dim0/1 = tile dims (no clipping),
//           tile_dim0/1, tensor_dim0_stride = ldb
// groups 2/3 zero (2-D tensor).  Tracked by TENSORcnt.
// ---------------------------------------------------------------------------
__device__ __forceinline__ void tdm_load_2d_f16(
    unsigned lds_off, const _Float16* gaddr, unsigned rows, unsigned cols,
    unsigned stride_elems)
{
  unsigned long long ga = (unsigned long long)(const void*)gaddr;
  v4u g0;
  g0[0] = 1u;                                  // count=1 (valid user descriptor)
  g0[1] = lds_off;                             // LDS byte address
  g0[2] = (unsigned)(ga & 0xFFFFFFFFull);      // global_addr[31:0]
  g0[3] = (unsigned)((ga >> 32) & 0x1FFFFFFull) | (2u << 30);  // [56:32] | type=2
  v8i g1;
  g1[0] = (int)(1u << 16);                     // wg_mask=0, data_size=1 (2 bytes)
  g1[1] = (int)((cols & 0xFFFFu) << 16);       // tensor_dim0[15:0]
  g1[2] = (int)(((cols >> 16) & 0xFFFFu) | ((rows & 0xFFFFu) << 16)); // td0 hi | td1 lo
  g1[3] = (int)(((rows >> 16) & 0xFFFFu) | ((cols & 0xFFFFu) << 16)); // td1 hi | tile_dim0
  g1[4] = (int)(rows & 0xFFFFu);               // tile_dim1 (tile_dim2 = 0)
  g1[5] = (int)stride_elems;                   // tensor_dim0_stride[31:0]
  g1[6] = 0;                                   // stride[47:32], dim1_stride lo
  g1[7] = 0;
  v4i z4 = {0, 0, 0, 0};
#if defined(__clang_major__) && __clang_major__ >= 23
  v8i z8 = {0, 0, 0, 0, 0, 0, 0, 0};
  __builtin_amdgcn_tensor_load_to_lds(g0, g1, z4, z4, z8, 0);
#else
  __builtin_amdgcn_tensor_load_to_lds(g0, g1, z4, z4, 0);
#endif
}

// ---------------------------------------------------------------------------
// WMMA GEMM: C[M,N] = A[M,K](f16) @ W[K,N](f16) + bias, optional bn/relu.
// Block = 128 threads (4 waves).  Block tile 128(M) x 64(N); each wave owns
// 32 M-rows.  B K-slabs (32x64) are TDM-DMA'd into double-buffered LDS by
// wave 0 and shared by all waves.  M%128==0, N%64==0, K%32==0.
// Layout assumptions (CDNA5 ISA 7.12.2):
//   A operand : lane half = K-block-of-8; a[0..7]=K(h*8..), a[8..15]=K(16+h*8..)
//   B operand : lane&15 = N column; b[0..7]=K(h*8..), b[8..15]=K(16+h*8..)
//   C/D       : M = v + 8*(lane>=16), N = lane&15
// ---------------------------------------------------------------------------
__global__ __launch_bounds__(128) void gemm16(
    const _Float16* __restrict__ A, const _Float16* __restrict__ W,
    int M, int N, int K, int lda, int ldb,
    const float* __restrict__ bias,
    const float* __restrict__ bng, const float* __restrict__ bnb,
    int relu,
    float* __restrict__ outF, int ldf,
    _Float16* __restrict__ outH, int ldh)
{
  __shared__ alignas(16) _Float16 lds_b[2][32][64];

  const int lane = threadIdx.x & 31;
  const int wv   = threadIdx.x >> 5;
  const int hf   = lane >> 4;
  const int mr   = lane & 15;
  const int m0   = (blockIdx.x * 4 + wv) * 32;
  const int n0   = blockIdx.y * 64;
  if (m0 >= M || n0 >= N) return;

  v8f acc[2][4];
#pragma unroll
  for (int mh = 0; mh < 2; ++mh)
#pragma unroll
    for (int j = 0; j < 4; ++j) acc[mh][j] = zero8();

  const _Float16* arow0 = A + (size_t)(m0 + mr) * lda + hf * 8;
  const _Float16* arow1 = arow0 + (size_t)16 * lda;

  // prologue: DMA first B slab into buffer 0
  if (wv == 0)
    tdm_load_2d_f16((unsigned)(size_t)&lds_b[0][0][0], W + n0, 32u, 64u,
                    (unsigned)ldb);

  int cur = 0;
  for (int k0 = 0; k0 < K; k0 += 32) {
    if (wv == 0) __builtin_amdgcn_s_wait_tensorcnt(0);
    __syncthreads();                       // slab `cur` is resident for all waves
    if (wv == 0 && k0 + 32 < K)            // prefetch next slab into other buffer
      tdm_load_2d_f16((unsigned)(size_t)&lds_b[cur ^ 1][0][0],
                      W + (size_t)(k0 + 32) * ldb + n0, 32u, 64u, (unsigned)ldb);

    HVec16 a0, a1;
    a0.half8[0] = *(const v8h*)(arow0 + k0);
    a0.half8[1] = *(const v8h*)(arow0 + k0 + 16);
    a1.half8[0] = *(const v8h*)(arow1 + k0);
    a1.half8[1] = *(const v8h*)(arow1 + k0 + 16);

#pragma unroll
    for (int j = 0; j < 4; ++j) {
      HVec16 b;
#pragma unroll
      for (int s = 0; s < 16; ++s) b.e[s] = lds_b[cur][(s < 8 ? hf * 8 + s : 8 + hf * 8 + s)][j * 16 + mr];
      acc[0][j] = wmma_acc(a0.v, b.v, acc[0][j]);
      acc[1][j] = wmma_acc(a1.v, b.v, acc[1][j]);
    }
    cur ^= 1;
  }

#pragma unroll
  for (int j = 0; j < 4; ++j) {
    const int col = n0 + j * 16 + mr;
    const float bs = bias ? bias[col] : 0.0f;
    float sc = 1.0f, sh = 0.0f;
    if (bng) { sc = bng[col] * rsqrtf(1.0f + EPS_); sh = bnb[col]; }
#pragma unroll
    for (int mh = 0; mh < 2; ++mh)
#pragma unroll
      for (int vv = 0; vv < 8; ++vv) {
        const int row = m0 + mh * 16 + vv + 8 * hf;
        float xv = acc[mh][j][vv] + bs;
        xv = xv * sc + sh;
        if (relu) xv = fmaxf(xv, 0.0f);
        if (outF) outF[(size_t)row * ldf + col] = xv;
        if (outH) outH[(size_t)row * ldh + col] = (_Float16)xv;
      }
  }
}

// ---------------------------------------------------------------------------
// Flash wavelet attention.  Q: (B,4096,256) f16, KV: (B,1024,512) f16
// (k cols [h*64..), v cols [256+h*64..)).  grid.x = b*4+h, grid.y = 64-row
// tile; block = 128 (4 waves x 16 q rows).  Writes cat(B,4096,320) cols [0,256).
// ---------------------------------------------------------------------------
__global__ __launch_bounds__(128) void wave_attn(
    const _Float16* __restrict__ Q, const _Float16* __restrict__ KV,
    _Float16* __restrict__ Ocat)
{
  const int lane = threadIdx.x & 31;
  const int wv   = threadIdx.x >> 5;
  const int hf   = lane >> 4;
  const int mr   = lane & 15;
  const int b    = blockIdx.x >> 2;
  const int h    = blockIdx.x & 3;
  const int m0   = blockIdx.y * 64 + wv * 16;

  __shared__ alignas(32) _Float16 lds_p[4][16][32];

  const _Float16* qb = Q  + (size_t)b * NTOK  * WDIM_       + h * HD_;
  const _Float16* kb = KV + (size_t)b * KVLEN * (2 * WDIM_) + h * HD_;
  const _Float16* vb = kb + WDIM_;

  const _Float16* qrow = qb + (size_t)(m0 + mr) * WDIM_ + hf * 8;
  HVec16 a0, a1;
  a0.half8[0] = *(const v8h*)(qrow);
  a0.half8[1] = *(const v8h*)(qrow + 16);
  a1.half8[0] = *(const v8h*)(qrow + 32);
  a1.half8[1] = *(const v8h*)(qrow + 48);

  float rmax[8], rsum[8];
  v8f oacc[4];
#pragma unroll
  for (int vv = 0; vv < 8; ++vv) { rmax[vv] = -3.0e38f; rsum[vv] = 0.0f; }
#pragma unroll
  for (int j = 0; j < 4; ++j) oacc[j] = zero8();

  for (int n0 = 0; n0 < KVLEN; n0 += 32) {
    // ---- S = Q @ K^T for 32 kv columns (two 16x16 tiles) ----
    v8f s0 = zero8(), s1 = zero8();
    {
      HVec16 bk;
#pragma unroll
      for (int e2 = 0; e2 < 16; ++e2) bk.e[e2] = kb[(size_t)(n0 + mr) * (2 * WDIM_) + (e2 < 8 ? hf * 8 + e2 : 8 + hf * 8 + e2)];
      s0 = wmma_acc(a0.v, bk.v, s0);
#pragma unroll
      for (int e2 = 0; e2 < 16; ++e2) bk.e[e2] = kb[(size_t)(n0 + mr) * (2 * WDIM_) + 32 + (e2 < 8 ? hf * 8 + e2 : 8 + hf * 8 + e2)];
      s0 = wmma_acc(a1.v, bk.v, s0);
#pragma unroll
      for (int e2 = 0; e2 < 16; ++e2) bk.e[e2] = kb[(size_t)(n0 + 16 + mr) * (2 * WDIM_) + (e2 < 8 ? hf * 8 + e2 : 8 + hf * 8 + e2)];
      s1 = wmma_acc(a0.v, bk.v, s1);
#pragma unroll
      for (int e2 = 0; e2 < 16; ++e2) bk.e[e2] = kb[(size_t)(n0 + 16 + mr) * (2 * WDIM_) + 32 + (e2 < 8 ? hf * 8 + e2 : 8 + hf * 8 + e2)];
      s1 = wmma_acc(a1.v, bk.v, s1);
    }
#pragma unroll
    for (int vv = 0; vv < 8; ++vv) { s0[vv] *= SCALE_; s1[vv] *= SCALE_; }

    // ---- online softmax (a row's 16 cols live across the lanes of a half) ----
    float cm[8];
#pragma unroll
    for (int vv = 0; vv < 8; ++vv) cm[vv] = fmaxf(s0[vv], s1[vv]);
#pragma unroll
    for (int d = 1; d < 16; d <<= 1)
#pragma unroll
      for (int vv = 0; vv < 8; ++vv) cm[vv] = fmaxf(cm[vv], __shfl_xor(cm[vv], d, 32));

    float corr[8], p0[8], p1[8], ps[8];
#pragma unroll
    for (int vv = 0; vv < 8; ++vv) {
      const float nm = fmaxf(rmax[vv], cm[vv]);
      corr[vv] = __expf(rmax[vv] - nm);
      rmax[vv] = nm;
      p0[vv] = __expf(s0[vv] - nm);
      p1[vv] = __expf(s1[vv] - nm);
      ps[vv] = p0[vv] + p1[vv];
    }
#pragma unroll
    for (int d = 1; d < 16; d <<= 1)
#pragma unroll
      for (int vv = 0; vv < 8; ++vv) ps[vv] += __shfl_xor(ps[vv], d, 32);
#pragma unroll
    for (int vv = 0; vv < 8; ++vv) rsum[vv] = rsum[vv] * corr[vv] + ps[vv];
#pragma unroll
    for (int j = 0; j < 4; ++j)
#pragma unroll
      for (int vv = 0; vv < 8; ++vv) oacc[j][vv] *= corr[vv];

    // ---- P: C-layout -> A-layout via LDS ----
#pragma unroll
    for (int vv = 0; vv < 8; ++vv) {
      const int row = vv + 8 * hf;
      lds_p[wv][row][mr]      = (_Float16)p0[vv];
      lds_p[wv][row][16 + mr] = (_Float16)p1[vv];
    }
    __syncthreads();
    HVec16 ap;
    const _Float16* pr = &lds_p[wv][mr][hf * 8];
    ap.half8[0] = *(const v8h*)(pr);
    ap.half8[1] = *(const v8h*)(pr + 16);
    __syncthreads();

    // ---- O += P(16x32) @ V(32x64) ----
#pragma unroll
    for (int j = 0; j < 4; ++j) {
      const _Float16* vrow = vb + (size_t)(n0 + lane) * (2 * WDIM_) + j * 16;
      HVec16 bv2;
#pragma unroll
      for (int s = 0; s < 16; ++s) bv2.e[s] = vb[(size_t)(n0 + (s < 8 ? hf * 8 + s : 8 + hf * 8 + s)) * (2 * WDIM_) + j * 16 + mr];
      oacc[j] = wmma_acc(ap.v, bv2.v, oacc[j]);
    }
  }

#pragma unroll
  for (int j = 0; j < 4; ++j)
#pragma unroll
    for (int vv = 0; vv < 8; ++vv) {
      const int row = m0 + vv + 8 * hf;
      const float o = oacc[j][vv] / rsum[vv];
      Ocat[((size_t)b * NTOK + row) * 320 + h * HD_ + j * 16 + mr] = (_Float16)o;
    }
}

// ---------------------------------------------------------------------------
// Window attention: 2x2 windows, 4 heads of 64; 4x4 attentions on VALU.
// qkv: (B,4096,768) f16 [q|k|v].  sa: (B,4096,256) f16.
// ---------------------------------------------------------------------------
__global__ void win_attn(const _Float16* __restrict__ qkv, _Float16* __restrict__ sa)
{
  const int tid = blockIdx.x * blockDim.x + threadIdx.x;
  if (tid >= B_ * 1024 * 4 * 4) return;
  const int l = tid & 3, h = (tid >> 2) & 3, g = (tid >> 4) & 1023, b = tid >> 14;
  const int gy = g >> 5, gx = g & 31;
  int toks[4];
#pragma unroll
  for (int dy = 0; dy < 2; ++dy)
#pragma unroll
    for (int dx = 0; dx < 2; ++dx)
      toks[dy * 2 + dx] = (2 * gy + dy) * WW_ + (2 * gx + dx);

  const _Float16* base = qkv + (size_t)b * NTOK * 768;
  const _Float16* qp = base + (size_t)toks[l] * 768 + h * HD_;
  float sc[4];
#pragma unroll
  for (int m = 0; m < 4; ++m) {
    const _Float16* kp = base + (size_t)toks[m] * 768 + 256 + h * HD_;
    float acc = 0.0f;
    for (int d = 0; d < HD_; ++d) acc += (float)qp[d] * (float)kp[d];
    sc[m] = acc * SCALE_;
  }
  const float mx = fmaxf(fmaxf(sc[0], sc[1]), fmaxf(sc[2], sc[3]));
  float p[4], s = 0.0f;
#pragma unroll
  for (int m = 0; m < 4; ++m) { p[m] = __expf(sc[m] - mx); s += p[m]; }
  const float inv = 1.0f / s;

  _Float16* op = sa + ((size_t)b * NTOK + toks[l]) * SDIM_ + h * HD_;
  for (int d = 0; d < HD_; ++d) {
    float o = 0.0f;
#pragma unroll
    for (int m = 0; m < 4; ++m)
      o += p[m] * (float)base[(size_t)toks[m] * 768 + 512 + h * HD_ + d];
    op[d] = (_Float16)(o * inv);
  }
}

// ---------------------------------------------------------------------------
// Elementwise helpers
// ---------------------------------------------------------------------------
__global__ void cast_f16_k(const float* __restrict__ in, _Float16* __restrict__ out, int n)
{
  const int i = blockIdx.x * blockDim.x + threadIdx.x;
  if (i < n) out[i] = (_Float16)in[i];
}

// Haar DWT: r (B,4096,64) f32 -> xdpre (B,1024,256) f16 (token-major, [ll|lh|hl|hh])
__global__ void dwt_k(const float* __restrict__ r, _Float16* __restrict__ xdpre)
{
  const int idx = blockIdx.x * blockDim.x + threadIdx.x;
  if (idx >= B_ * 1024 * RED_) return;
  const int c = idx & 63, p = (idx >> 6) & 1023, b = idx >> 16;
  const int y = p >> 5, x = p & 31;
  const size_t base = (size_t)b * NTOK * RED_;
  const float a  = r[base + (size_t)((2 * y) * WW_ + 2 * x) * RED_ + c];
  const float bv = r[base + (size_t)((2 * y) * WW_ + 2 * x + 1) * RED_ + c];
  const float cv = r[base + (size_t)((2 * y + 1) * WW_ + 2 * x) * RED_ + c];
  const float dv = r[base + (size_t)((2 * y + 1) * WW_ + 2 * x + 1) * RED_ + c];
  const size_t o = ((size_t)b * 1024 + p) * 256;
  xdpre[o + c]       = (_Float16)(0.5f * (a + bv + cv + dv));
  xdpre[o + 64 + c]  = (_Float16)(0.5f * (a + bv - cv - dv));
  xdpre[o + 128 + c] = (_Float16)(0.5f * (a - bv + cv - dv));
  xdpre[o + 192 + c] = (_Float16)(0.5f * (a - bv - cv + dv));
}

// iDWT: xd (B,1024,256) f32 -> cat (B,4096,320) cols [256,320)
__global__ void idwt_k(const float* __restrict__ xd, _Float16* __restrict__ cat)
{
  const int idx = blockIdx.x * blockDim.x + threadIdx.x;
  if (idx >= B_ * 1024 * RED_) return;
  const int c = idx & 63, p = (idx >> 6) & 1023, b = idx >> 16;
  const size_t o = ((size_t)b * 1024 + p) * 256;
  const float ll = xd[o + c], lh = xd[o + 64 + c], hl = xd[o + 128 + c], hh = xd[o + 192 + c];
  const float y00 = 0.5f * (ll + lh + hl + hh);
  const float y01 = 0.5f * (ll + lh - hl - hh);
  const float y10 = 0.5f * (ll - lh + hl - hh);
  const float y11 = 0.5f * (ll - lh - hl + hh);
  const int y = p >> 5, x = p & 31;
  const size_t cb = (size_t)b * NTOK * 320 + 256 + c;
  cat[cb + (size_t)((2 * y) * WW_ + 2 * x) * 320]         = (_Float16)y00;
  cat[cb + (size_t)((2 * y) * WW_ + 2 * x + 1) * 320]     = (_Float16)y01;
  cat[cb + (size_t)((2 * y + 1) * WW_ + 2 * x) * 320]     = (_Float16)y10;
  cat[cb + (size_t)((2 * y + 1) * WW_ + 2 * x + 1) * 320] = (_Float16)y11;
}

// filter_w (O=256, I=256, 3, 3) -> W2[(t*256+i), o]
__global__ void repack_filter_k(const float* __restrict__ fw, _Float16* __restrict__ w2)
{
  const int idx = blockIdx.x * blockDim.x + threadIdx.x;
  if (idx >= 256 * 256 * 9) return;
  const int t = idx % 9, i = (idx / 9) % 256, o = idx / (9 * 256);
  w2[((size_t)t * 256 + i) * 256 + o] = (_Float16)fw[((size_t)o * 256 + i) * 9 + t];
}

// im2col, SAME padding: xdpre (B,1024,256) f16 -> col (B,1024,2304) f16
__global__ void im2col_k(const _Float16* __restrict__ xdpre, _Float16* __restrict__ col)
{
  const int idx = blockIdx.x * blockDim.x + threadIdx.x;
  if (idx >= B_ * 1024 * 2304) return;
  const int c2 = idx % 2304;
  const int p  = (idx / 2304) % 1024;
  const int b  = idx / (2304 * 1024);
  const int t = c2 / 256, ci = c2 % 256;
  const int ky = t / 3, kx = t % 3;
  const int y = (p >> 5) + ky - 1, x = (p & 31) + kx - 1;
  _Float16 v = (_Float16)0.0f;
  if (y >= 0 && y < 32 && x >= 0 && x < 32)
    v = xdpre[((size_t)b * 1024 + y * 32 + x) * 256 + ci];
  col[(size_t)idx] = v;
}

// LayerNorm over 256 channels per position: xd (B,1024,256) f32 -> kvin f16
__global__ __launch_bounds__(256) void lnorm_k(
    const float* __restrict__ xd, const float* __restrict__ g,
    const float* __restrict__ bb, _Float16* __restrict__ kvin)
{
  __shared__ float red[256];
  const int row = blockIdx.x, c = threadIdx.x;
  const float v = xd[(size_t)row * 256 + c];
  red[c] = v;
  __syncthreads();
  for (int s = 128; s > 0; s >>= 1) { if (c < s) red[c] += red[c + s]; __syncthreads(); }
  const float mean = red[0] / 256.0f;
  __syncthreads();
  const float d = v - mean;
  red[c] = d * d;
  __syncthreads();
  for (int s = 128; s > 0; s >>= 1) { if (c < s) red[c] += red[c + s]; __syncthreads(); }
  const float var = red[0] / 256.0f;
  const float y = d * rsqrtf(var + EPS_) * g[c] + bb[c];
  kvin[(size_t)row * 256 + c] = (_Float16)y;
}

// ---------------------------------------------------------------------------
extern "C" void kernel_launch(void* const* d_in, const int* in_sizes, int n_in,
                              void* d_out, int out_size, void* d_ws, size_t ws_size,
                              hipStream_t stream)
{
  (void)in_sizes; (void)n_in; (void)out_size;
  const float* x            = (const float*)d_in[0];
  const float* wq_w         = (const float*)d_in[1];
  const float* kv_ln_g      = (const float*)d_in[2];
  const float* kv_ln_b      = (const float*)d_in[3];
  const float* wkv_w        = (const float*)d_in[4];
  const float* wkv_b        = (const float*)d_in[5];
  const float* wproj_w      = (const float*)d_in[6];
  const float* wproj_b      = (const float*)d_in[7];
  const float* reduce_w     = (const float*)d_in[8];
  const float* reduce_b     = (const float*)d_in[9];
  const float* reduce_bn_g  = (const float*)d_in[10];
  const float* reduce_bn_b  = (const float*)d_in[11];
  const float* filter_w     = (const float*)d_in[12];
  const float* filter_b     = (const float*)d_in[13];
  const float* filter_bn_g  = (const float*)d_in[14];
  const float* filter_bn_b  = (const float*)d_in[15];
  const float* sqkv_w       = (const float*)d_in[16];
  const float* sproj_w      = (const float*)d_in[17];
  const float* sproj_b      = (const float*)d_in[18];
  float* out = (float*)d_out;

  char* ws = (char*)d_ws;
  size_t off = 0;
  auto alloc = [&](size_t bytes) -> char* {
    char* p = ws + off;
    off = (off + bytes + 255) & ~(size_t)255;
    return p;
  };

  _Float16* xf      = (_Float16*)alloc((size_t)NTOK * B_ * DIM_ * 2);
  _Float16* sqkvF   = (_Float16*)alloc((size_t)DIM_ * 768 * 2);
  _Float16* wqF     = (_Float16*)alloc((size_t)DIM_ * WDIM_ * 2);
  _Float16* redF    = (_Float16*)alloc((size_t)DIM_ * RED_ * 2);
  _Float16* wkvF    = (_Float16*)alloc((size_t)WDIM_ * 512 * 2);
  _Float16* sprojF  = (_Float16*)alloc((size_t)SDIM_ * SDIM_ * 2);
  _Float16* wprojF  = (_Float16*)alloc((size_t)320 * WDIM_ * 2);
  _Float16* w2F     = (_Float16*)alloc((size_t)2304 * 256 * 2);
  _Float16* qkvsF   = (_Float16*)alloc((size_t)B_ * NTOK * 768 * 2);
  _Float16* qF      = (_Float16*)alloc((size_t)B_ * NTOK * WDIM_ * 2);
  float*    rF32    = (float*)   alloc((size_t)B_ * NTOK * RED_ * 4);
  _Float16* xdpreF  = (_Float16*)alloc((size_t)B_ * 1024 * 256 * 2);
  _Float16* colF    = (_Float16*)alloc((size_t)B_ * 1024 * 2304 * 2);
  float*    xdF32   = (float*)   alloc((size_t)B_ * 1024 * 256 * 4);
  _Float16* kvinF   = (_Float16*)alloc((size_t)B_ * 1024 * 256 * 2);
  _Float16* kvfF    = (_Float16*)alloc((size_t)B_ * 1024 * 512 * 2);
  _Float16* saF     = (_Float16*)alloc((size_t)B_ * NTOK * SDIM_ * 2);
  _Float16* catF    = (_Float16*)alloc((size_t)B_ * NTOK * 320 * 2);
  if (off > ws_size) return;  // workspace too small; nothing safe to do

  const int TPB = 256;
  auto nb = [&](int n) { return (n + TPB - 1) / TPB; };

  // ---- f32 -> f16 casts ----
  cast_f16_k<<<nb(B_ * NTOK * DIM_), TPB, 0, stream>>>(x,        xf,     B_ * NTOK * DIM_);
  cast_f16_k<<<nb(DIM_ * 768),       TPB, 0, stream>>>(sqkv_w,   sqkvF,  DIM_ * 768);
  cast_f16_k<<<nb(DIM_ * WDIM_),     TPB, 0, stream>>>(wq_w,     wqF,    DIM_ * WDIM_);
  cast_f16_k<<<nb(DIM_ * RED_),      TPB, 0, stream>>>(reduce_w, redF,   DIM_ * RED_);
  cast_f16_k<<<nb(WDIM_ * 512),      TPB, 0, stream>>>(wkv_w,    wkvF,   WDIM_ * 512);
  cast_f16_k<<<nb(SDIM_ * SDIM_),    TPB, 0, stream>>>(sproj_w,  sprojF, SDIM_ * SDIM_);
  cast_f16_k<<<nb(320 * WDIM_),      TPB, 0, stream>>>(wproj_w,  wprojF, 320 * WDIM_);
  repack_filter_k<<<nb(256 * 256 * 9), TPB, 0, stream>>>(filter_w, w2F);

  const int M = B_ * NTOK;  // 16384 token rows

  // ---- qkv_s = x @ sqkv_w  (f16 out) ----
  gemm16<<<dim3(M / 128, 768 / 64), 128, 0, stream>>>(
      xf, sqkvF, M, 768, DIM_, DIM_, 768,
      nullptr, nullptr, nullptr, 0, nullptr, 0, qkvsF, 768);

  // ---- q = x @ wq_w  (f16 out) ----
  gemm16<<<dim3(M / 128, WDIM_ / 64), 128, 0, stream>>>(
      xf, wqF, M, WDIM_, DIM_, DIM_, WDIM_,
      nullptr, nullptr, nullptr, 0, nullptr, 0, qF, WDIM_);

  // ---- r = relu(bn(x @ reduce_w + reduce_b))  (f32 out) ----
  gemm16<<<dim3(M / 128, 1), 128, 0, stream>>>(
      xf, redF, M, RED_, DIM_, DIM_, RED_,
      reduce_b, reduce_bn_g, reduce_bn_b, 1, rF32, RED_, nullptr, 0);

  // ---- Haar DWT ----
  dwt_k<<<nb(B_ * 1024 * RED_), TPB, 0, stream>>>(rF32, xdpreF);

  // ---- im2col + conv GEMM: xd = relu(bn(conv3x3(xd_pre) + filter_b)) ----
  im2col_k<<<nb(B_ * 1024 * 2304), TPB, 0, stream>>>(xdpreF, colF);
  gemm16<<<dim3((B_ * 1024) / 128, 256 / 64), 128, 0, stream>>>(
      colF, w2F, B_ * 1024, 256, 2304, 2304, 256,
      filter_b, filter_bn_g, filter_bn_b, 1, xdF32, 256, nullptr, 0);

  // ---- iDWT -> concat buffer cols [256,320) ----
  idwt_k<<<nb(B_ * 1024 * RED_), TPB, 0, stream>>>(xdF32, catF);

  // ---- layernorm(kv_in) ----
  lnorm_k<<<B_ * 1024, 256, 0, stream>>>(xdF32, kv_ln_g, kv_ln_b, kvinF);

  // ---- kv = kv_in @ wkv_w + wkv_b  (f16 out) ----
  gemm16<<<dim3((B_ * 1024) / 128, 512 / 64), 128, 0, stream>>>(
      kvinF, wkvF, B_ * 1024, 512, WDIM_, WDIM_, 512,
      wkv_b, nullptr, nullptr, 0, nullptr, 0, kvfF, 512);

  // ---- window self-attention (tiny 4x4 -> VALU) ----
  win_attn<<<nb(B_ * 1024 * 16), TPB, 0, stream>>>(qkvsF, saF);

  // ---- wavelet cross-attention (flash, WMMA) -> concat cols [0,256) ----
  wave_attn<<<dim3(B_ * 4, NTOK / 64), 128, 0, stream>>>(qF, kvfF, catF);

  // ---- sa_out = o @ sproj_w + sproj_b -> out cols [0,256) ----
  gemm16<<<dim3(M / 128, SDIM_ / 64), 128, 0, stream>>>(
      saF, sprojF, M, SDIM_, SDIM_, SDIM_, SDIM_,
      sproj_b, nullptr, nullptr, 0, out, DIM_, nullptr, 0);

  // ---- wa_out = cat @ wproj_w + wproj_b -> out cols [256,512) ----
  gemm16<<<dim3(M / 128, WDIM_ / 64), 128, 0, stream>>>(
      catF, wprojF, M, WDIM_, 320, 320, WDIM_,
      wproj_b, nullptr, nullptr, 0, out + 256, DIM_, nullptr, 0);
}
